// SparseMoE_84146999263306
// MI455X (gfx1250) — matrix operand order repaired and checked by reading, not compiled, not run on hardware
//
#include <hip/hip_runtime.h>
#include <hip/hip_bf16.h>
#include <math.h>

#define D_MODEL 1024
#define HIDDEN  4096
#define NEXP    8
#define NTOK    4096   // BATCH*SEQ
#define MT      32     // tokens per block
#define HC      64     // hidden chunk per iteration (two 32-row TDM stages)

typedef _Float16 f16;
typedef f16   v16h __attribute__((ext_vector_type(16)));
typedef f16   v4h  __attribute__((ext_vector_type(4)));
typedef float v8f  __attribute__((ext_vector_type(8)));
typedef unsigned int v4u __attribute__((ext_vector_type(4)));
typedef int   v8i  __attribute__((ext_vector_type(8)));
typedef int   v4i  __attribute__((ext_vector_type(4)));

__device__ __forceinline__ v8f wmma16(v16h a, v16h b, v8f c) {
    // -> v_wmma_f32_16x16x32_f16
    return __builtin_amdgcn_wmma_f32_16x16x32_f16(false, a, false, b, (short)0, c, false, false);
}

// A fragment (16x32 f16, ISA 7.12.2): lane holds row m = lane&15.
// element i -> k = k0 + i + (lane>=16 ? 16 : 0) (this code's contiguous order, same as B; not the ISA table order)
__device__ __forceinline__ v16h load_a_lds(const f16* base, int row, int stride,
                                           int k0, int lane) {
    const f16* p = base + row * stride + k0 + ((lane & 16) ? 16 : 0);
    v16h a;
#pragma unroll
    for (int i = 0; i < 16; ++i) a[i] = p[i];
    return a;
}

// B fragment (32x16 f16) from row-major f32 weights W[k][n], leading dim ldn (global).
__device__ __forceinline__ v16h load_b_w(const float* __restrict__ base, int k0,
                                         int ldn, int col0, int lane) {
    const float* p = base + (size_t)(k0 + ((lane & 16) ? 16 : 0)) * ldn
                          + col0 + (lane & 15);
    v16h b;
#pragma unroll
    for (int i = 0; i < 16; ++i) b[i] = (f16)p[(size_t)i * ldn];
    return b;
}

// B fragment from an LDS-staged f32 chunk [32][ldn]
__device__ __forceinline__ v16h load_b_lds32(const float* base, int ldn, int col0, int lane) {
    const float* p = base + ((lane & 16) ? 16 : 0) * ldn + col0 + (lane & 15);
    v16h b;
#pragma unroll
    for (int i = 0; i < 16; ++i) b[i] = (f16)p[i * ldn];
    return b;
}

// ---- Tensor Data Mover: DMA a [32 x 1024] f32 tile (row stride 1024) into LDS ----
__device__ __forceinline__ void tdm_load_w2_chunk(const float* gptr, unsigned lds_off) {
    unsigned long long ga = (unsigned long long)(uintptr_t)gptr;
    v4u g0;
    g0[0] = 1u;                                   // count=1 valid descriptor
    g0[1] = lds_off;                              // lds_addr (bytes)
    g0[2] = (unsigned)(ga & 0xffffffffu);         // global_addr[31:0]
    g0[3] = (unsigned)((ga >> 32) & 0x1ffffffu)   // global_addr[56:32]
            | (2u << 30);                         // type=2 ("image")
    v8i g1;
    g1[0] = (int)(2u << 16);                      // data_size=2 -> 4 bytes
    g1[1] = (int)((1024u & 0xffffu) << 16);       // tensor_dim0[15:0]
    g1[2] = (int)((1024u >> 16) | ((4096u & 0xffffu) << 16)); // dim0 hi | tensor_dim1 lo
    g1[3] = (int)((4096u >> 16) | (1024u << 16)); // tensor_dim1 hi | tile_dim0=1024
    g1[4] = 32;                                   // tile_dim1=32 rows, tile_dim2=0
    g1[5] = 1024;                                 // tensor_dim0_stride[31:0]
    g1[6] = 0;                                    // stride hi | tensor_dim1_stride lo
    g1[7] = 0;
    v4i z4 = {};
#if __clang_major__ >= 23
    v8i z8 = {};
    __builtin_amdgcn_tensor_load_to_lds(g0, g1, z4, z4, z8, 0);
#else
    __builtin_amdgcn_tensor_load_to_lds(g0, g1, z4, z4, 0);
#endif
}

// ---------------- Kernel 1: gating (softmax + top-2 -> comb[N,8]) ----------
__global__ void gate_kernel(const float* __restrict__ x, const float* __restrict__ Wg,
                            const float* __restrict__ bg, float* __restrict__ comb) {
    __shared__ float sWg[D_MODEL * NEXP];   // 32 KB
    __shared__ float sbg[NEXP];
    const int tid = threadIdx.x;
    for (int i = tid; i < D_MODEL * NEXP; i += blockDim.x) sWg[i] = Wg[i];
    if (tid < NEXP) sbg[tid] = bg[tid];
    __syncthreads();

    const int t = blockIdx.x * blockDim.x + tid;
    if (t >= NTOK) return;
    float acc[NEXP];
#pragma unroll
    for (int e = 0; e < NEXP; ++e) acc[e] = sbg[e];
    const float* xr = x + (size_t)t * D_MODEL;
    for (int d = 0; d < D_MODEL; ++d) {
        float xv = xr[d];
#pragma unroll
        for (int e = 0; e < NEXP; ++e) acc[e] += xv * sWg[d * NEXP + e];
    }
    float m = acc[0];
#pragma unroll
    for (int e = 1; e < NEXP; ++e) m = fmaxf(m, acc[e]);
    float g[NEXP], s = 0.0f;
#pragma unroll
    for (int e = 0; e < NEXP; ++e) { g[e] = expf(acc[e] - m); s += g[e]; }
    const float inv = 1.0f / s;
    int i1 = 0; float v1 = g[0];
#pragma unroll
    for (int e = 1; e < NEXP; ++e) if (g[e] > v1) { v1 = g[e]; i1 = e; }
    int i2 = -1; float v2 = -1.0f;
#pragma unroll
    for (int e = 0; e < NEXP; ++e) if (e != i1 && g[e] > v2) { v2 = g[e]; i2 = e; }
#pragma unroll
    for (int e = 0; e < NEXP; ++e)
        comb[(size_t)t * NEXP + e] = (e == i1) ? v1 * inv : (e == i2) ? v2 * inv : 0.0f;
}

// ---------------- Kernel 2: out = sum_e comb[t,e]*b2[e,:] (init/clear) -----
__global__ void init_out_kernel(const float* __restrict__ comb,
                                const float* __restrict__ b2,
                                float* __restrict__ out) {
    const int idx = blockIdx.x * blockDim.x + threadIdx.x;
    const int t = idx >> 10;
    const int d = idx & (D_MODEL - 1);
    float s = 0.0f;
#pragma unroll
    for (int e = 0; e < NEXP; ++e)
        s += comb[(size_t)t * NEXP + e] * b2[(size_t)e * D_MODEL + d];
    out[idx] = s;
}

// ---------------- Kernel 3: fused expert FFN (WMMA + TDM) ------------------
// grid = (NTOK/MT, NEXP); 256 threads = 8 waves.
__global__ void moe_ffn_kernel(const float* __restrict__ x,
                               const float* __restrict__ W1,
                               const float* __restrict__ b1,
                               const float* __restrict__ W2,
                               const float* __restrict__ comb,
                               float* __restrict__ out) {
    extern __shared__ char smem[];
    f16*   s_x    = (f16*)smem;                                 // [MT][D_MODEL] 64KB
    f16*   s_h    = (f16*)(smem + MT * D_MODEL * 2);            // [MT][HC]       4KB
    float* s_w2   = (float*)(smem + MT * D_MODEL * 2 + MT * HC * 2); // [32][1024] 128KB
    float* s_comb = (float*)((char*)s_w2 + 32 * D_MODEL * 4);
    int*   s_any  = (int*)(s_comb + MT);

    const int tile = blockIdx.x;
    const int e    = blockIdx.y;
    const int tid  = threadIdx.x;
    const int lane = tid & 31;
    const int w    = tid >> 5;

    if (tid == 0) *s_any = 0;
    __syncthreads();
    if (tid < MT) {
        float c = comb[(size_t)(tile * MT + tid) * NEXP + e];
        s_comb[tid] = c;
        if (c != 0.0f) atomicOr(s_any, 1);
    }
    __syncthreads();
    if (*s_any == 0) return;   // block-uniform: EXEC stays all-ones for WMMA/TDM

    // stage x tile (f32 -> f16) into LDS, float4-wide
    for (int idx = tid; idx < MT * D_MODEL / 4; idx += blockDim.x) {
        int r = idx >> 8, c4 = (idx & 255) * 4;
        const float4 v = *(const float4*)&x[(size_t)(tile * MT + r) * D_MODEL + c4];
        v4h h4; h4[0] = (f16)v.x; h4[1] = (f16)v.y; h4[2] = (f16)v.z; h4[3] = (f16)v.w;
        *(v4h*)&s_x[r * D_MODEL + c4] = h4;
    }

    const float* W1e = W1 + (size_t)e * D_MODEL * HIDDEN;
    const float* W2e = W2 + (size_t)e * HIDDEN * D_MODEL;
    const float* b1e = b1 + (size_t)e * HIDDEN;

    const int mh   = w >> 2;              // token half: rows mh*16..mh*16+15
    const int ht   = w & 3;               // GEMM1: 16-col tile inside HC chunk
    const int nq   = w & 3;               // GEMM2: 256-col quarter of D_MODEL
    const int rowA = mh * 16 + (lane & 15);
    const int rhi  = (lane & 16) ? 8 : 0;
    const unsigned w2_off = (unsigned)(uintptr_t)s_w2;   // LDS byte offset

    // prime the pipeline: DMA W2 rows [0,32) while we stage x / run GEMM1
    if (w == 0) tdm_load_w2_chunk(W2e, w2_off);
    __syncthreads();

    v8f acc[16] = {};                     // 32x1024 block output, in VGPRs

    for (int hc = 0; hc < HIDDEN; hc += HC) {
        // ---- GEMM1: h_tile = x_tile @ W1[:, hc+ht*16 .. +16], K = D_MODEL
        v8f hv = {};
        const int hcol = hc + ht * 16;
#pragma unroll 4
        for (int k0 = 0; k0 < D_MODEL; k0 += 32) {
            v16h a = load_a_lds(s_x, rowA, D_MODEL, k0, lane);
            v16h b = load_b_w(W1e, k0, HIDDEN, hcol, lane);
            hv = wmma16(a, b, hv);
        }
        if (hc + HC < HIDDEN)
            __builtin_prefetch((const void*)(W1e + (size_t)lane * HIDDEN + hc + HC), 0, 1);

        // bias + exact GELU + fold per-token gate weight into h rows
        const float b1v = b1e[hcol + (lane & 15)];
#pragma unroll
        for (int r = 0; r < 8; ++r) {
            const int tr = mh * 16 + r + rhi;
            float v = hv[r] + b1v;
            float gl = 0.5f * v * (1.0f + erff(v * 0.70710678118654752f));
            s_h[tr * HC + ht * 16 + (lane & 15)] = (f16)(gl * s_comb[tr]);
        }
        if (w == 0) __builtin_amdgcn_s_wait_tensorcnt(0);  // W2 rows [hc,hc+32) landed
        __syncthreads();                                   // also publishes s_h

        // ---- GEMM2 k-half 0: acc += h[:,0:32] @ s_w2
        v16h a2_0 = load_a_lds(s_h, rowA, HC, 0, lane);
#pragma unroll
        for (int nt = 0; nt < 16; ++nt) {
            v16h b0 = load_b_lds32(s_w2, D_MODEL, nq * 256 + nt * 16, lane);
            acc[nt] = wmma16(a2_0, b0, acc[nt]);
        }
        __syncthreads();                                   // everyone done with s_w2

        // ---- stage + consume k-half 1: W2 rows [hc+32, hc+64)
        if (w == 0) {
            tdm_load_w2_chunk(W2e + (size_t)(hc + 32) * D_MODEL, w2_off);
            __builtin_amdgcn_s_wait_tensorcnt(0);
        }
        __syncthreads();
        v16h a2_1 = load_a_lds(s_h, rowA, HC, 32, lane);
#pragma unroll
        for (int nt = 0; nt < 16; ++nt) {
            v16h b1f = load_b_lds32(s_w2, D_MODEL, nq * 256 + nt * 16, lane);
            acc[nt] = wmma16(a2_1, b1f, acc[nt]);
        }
        __syncthreads();                                   // retire s_w2 + s_h readers

        // overlap next chunk's k-half-0 DMA with the next GEMM1
        if (w == 0 && hc + HC < HIDDEN)
            tdm_load_w2_chunk(W2e + (size_t)(hc + HC) * D_MODEL, w2_off);
    }

    // merge experts into output (f32 atomics; out pre-initialized with b2 term)
#pragma unroll
    for (int nt = 0; nt < 16; ++nt) {
        const int col = nq * 256 + nt * 16 + (lane & 15);
#pragma unroll
        for (int r = 0; r < 8; ++r) {
            const int tok = tile * MT + mh * 16 + r + rhi;
            atomicAdd(&out[(size_t)tok * D_MODEL + col], acc[nt][r]);
        }
    }
}

extern "C" void kernel_launch(void* const* d_in, const int* in_sizes, int n_in,
                              void* d_out, int out_size, void* d_ws, size_t ws_size,
                              hipStream_t stream) {
    const float* x  = (const float*)d_in[0];
    const float* Wg = (const float*)d_in[1];
    const float* bg = (const float*)d_in[2];
    const float* W1 = (const float*)d_in[3];
    const float* b1 = (const float*)d_in[4];
    const float* W2 = (const float*)d_in[5];
    const float* b2 = (const float*)d_in[6];
    float* out  = (float*)d_out;
    float* comb = (float*)d_ws;   // [NTOK][NEXP] f32 = 128 KB

    gate_kernel<<<NTOK / 256, 256, 0, stream>>>(x, Wg, bg, comb);
    init_out_kernel<<<(NTOK * D_MODEL) / 256, 256, 0, stream>>>(comb, b2, out);

    dim3 grid(NTOK / MT, NEXP);
    size_t smem = (size_t)MT * D_MODEL * 2    // s_x
                + (size_t)MT * HC * 2         // s_h
                + (size_t)32 * D_MODEL * 4    // s_w2 stage
                + MT * 4 + 16;                // s_comb, s_any
    moe_ffn_kernel<<<grid, 256, smem, stream>>>(x, W1, b1, W2, comb, out);
}
